// LuminaRegimeDetector_48576080118316
// MI455X (gfx1250) — matrix operand order repaired and checked by reading, not compiled, not run on hardware
//
#include <hip/hip_runtime.h>
#include <stdint.h>

#define B_ 16
#define T_ 1024
#define NF_ 64
#define D_ 128
#define H_ 4
#define DH_ 32
#define DFF_ 512
#define R_ 4
#define L_ 4
#define BT_ (B_*T_)

typedef __attribute__((ext_vector_type(16))) _Float16 v16h;
typedef __attribute__((ext_vector_type(8)))  _Float16 v8h;
typedef __attribute__((ext_vector_type(8)))  float    v8f;

// A-operand (16x32 f16): halves 0..7 <- k = grp*8 + j ; halves 8..15 <- k = 16 + grp*8 + (j-8)
__device__ __forceinline__ v16h load_afrag(const _Float16* __restrict__ row, int kk, int grp) {
  v8h lo = *(const v8h*)(row + kk + grp * 8);
  v8h hi = *(const v8h*)(row + kk + 16 + grp * 8);
  return __builtin_shufflevector(lo, hi, 0, 1, 2, 3, 4, 5, 6, 7,
                                 8, 9, 10, 11, 12, 13, 14, 15);
}
// B-operand (32x16 f16): halves 0..7 <- k = grp*8 + j ; halves 8..15 <- k = 16 + grp*8 + (j-8)
__device__ __forceinline__ v16h load_bfrag(const _Float16* __restrict__ row, int kk, int grp) {
  v8h lo = *(const v8h*)(row + kk + grp * 8);
  v8h hi = *(const v8h*)(row + kk + 16 + grp * 8);
  return __builtin_shufflevector(lo, hi, 0, 1, 2, 3, 4, 5, 6, 7,
                                 8, 9, 10, 11, 12, 13, 14, 15);
}

// Async global -> LDS 16B copy per lane (ASYNCcnt-tracked), gfx1250 VGLOBAL op 98.
__device__ __forceinline__ void async_b128_to_lds(const _Float16* gsrc, _Float16* lds_dst) {
  unsigned loff = (unsigned)(uintptr_t)lds_dst;          // low 32 bits = LDS byte addr
  unsigned long long gaddr = (unsigned long long)(uintptr_t)gsrc;
  asm volatile("global_load_async_to_lds_b128 %0, %1, off"
               :: "v"(loff), "v"(gaddr) : "memory");
}
__device__ __forceinline__ void wait_async0() {
  asm volatile("s_wait_asynccnt 0x0" ::: "memory");
}

// ---------------------------------------------------------------------------
// f32 -> f16 pack
// ---------------------------------------------------------------------------
__global__ __launch_bounds__(256)
void pack_f16(const float* __restrict__ src, _Float16* __restrict__ dst, int n) {
  int i = blockIdx.x * 256 + threadIdx.x;
  if (i < n) dst[i] = (_Float16)src[i];
}

// ---------------------------------------------------------------------------
// GEMM (f16 in, f32 acc): C[M,N] = act(A[M,K] @ W[N,K]^T + bias [+ resid])
// mode: 0 = plain, 1 = relu, 2 = add resid. C (f32) and C16 (f16) optional.
// ---------------------------------------------------------------------------
__global__ __launch_bounds__(128)
void gemm16(const _Float16* __restrict__ A, const _Float16* __restrict__ W,
            const float* __restrict__ bias, const float* __restrict__ resid,
            float* __restrict__ C, _Float16* __restrict__ C16,
            int M, int N, int K, int mode) {
  int wid  = threadIdx.x >> 5;
  int lane = threadIdx.x & 31;
  int tile = blockIdx.x * 4 + wid;
  int ntn  = N >> 4;
  int total = (M >> 4) * ntn;
  if (tile >= total) return;
  int mt = tile / ntn, nt = tile % ntn;
  int grp = lane >> 4, idx = lane & 15;

  const _Float16* arow = A + (size_t)(mt * 16 + idx) * K;
  const _Float16* wrow = W + (size_t)(nt * 16 + idx) * K;

  v8f acc = {};
  for (int kk = 0; kk < K; kk += 32) {
    __builtin_prefetch(arow + kk + 32, 0, 1);
    __builtin_prefetch(wrow + kk + 32, 0, 1);
    v16h af = load_afrag(arow, kk, grp);
    v16h bf = load_bfrag(wrow, kk, grp);
    acc = __builtin_amdgcn_wmma_f32_16x16x32_f16(false, af, false, bf,
                                                 (short)0, acc, false, false);
  }

  int n = nt * 16 + idx;
  float bval = bias ? bias[n] : 0.f;
#pragma unroll
  for (int r = 0; r < 8; ++r) {
    int m = mt * 16 + r + grp * 8;
    float val = acc[r] + bval;
    if (mode == 1) val = fmaxf(val, 0.f);
    if (mode == 2) val += resid[(size_t)m * N + n];
    if (C)   C[(size_t)m * N + n]   = val;
    if (C16) C16[(size_t)m * N + n] = (_Float16)val;
  }
}

// ---------------------------------------------------------------------------
// QKV GEMM (f16 in): scatter q/k -> [B,H,T,DH] f16, v -> [B,H,DH,T] f16.
// ---------------------------------------------------------------------------
__global__ __launch_bounds__(128)
void gemm_qkv16(const _Float16* __restrict__ A, const _Float16* __restrict__ W,
                const float* __restrict__ bias,
                _Float16* __restrict__ Qh, _Float16* __restrict__ Kh,
                _Float16* __restrict__ VhT) {
  int wid  = threadIdx.x >> 5;
  int lane = threadIdx.x & 31;
  int tile = blockIdx.x * 4 + wid;
  const int ntn = (3 * D_) >> 4;          // 24
  int total = (BT_ >> 4) * ntn;
  if (tile >= total) return;
  int mt = tile / ntn, nt = tile % ntn;
  int grp = lane >> 4, idx = lane & 15;

  const _Float16* arow = A + (size_t)(mt * 16 + idx) * D_;
  const _Float16* wrow = W + (size_t)(nt * 16 + idx) * D_;

  v8f acc = {};
  for (int kk = 0; kk < D_; kk += 32) {
    v16h af = load_afrag(arow, kk, grp);
    v16h bf = load_bfrag(wrow, kk, grp);
    acc = __builtin_amdgcn_wmma_f32_16x16x32_f16(false, af, false, bf,
                                                 (short)0, acc, false, false);
  }

  int n = nt * 16 + idx;
  float bval = bias[n];
  int sect   = n >> 7;          // 0=q 1=k 2=v
  int within = n & 127;
  int head   = within >> 5;
  int dh     = within & 31;
#pragma unroll
  for (int r = 0; r < 8; ++r) {
    int m = mt * 16 + r + grp * 8;
    int b = m >> 10, t = m & (T_ - 1);
    int bh = b * H_ + head;
    _Float16 val = (_Float16)(acc[r] + bval);
    if (sect == 0)      Qh[((size_t)bh * T_ + t) * DH_ + dh] = val;
    else if (sect == 1) Kh[((size_t)bh * T_ + t) * DH_ + dh] = val;
    else                VhT[((size_t)bh * DH_ + dh) * T_ + t] = val;
  }
}

// ---------------------------------------------------------------------------
// Flash attention. One block = one (b,h) x 4 query tiles (4 waves).
// K/V chunks (32 keys) are staged into LDS with async global->LDS loads,
// double-buffered so the async DMA overlaps the WMMA + softmax work.
// ---------------------------------------------------------------------------
__global__ __launch_bounds__(128)
void attn_kernel(const _Float16* __restrict__ Qh, const _Float16* __restrict__ Kh,
                 const _Float16* __restrict__ VhT, _Float16* __restrict__ attn16) {
  __shared__ alignas(64) _Float16 pbuf[4][16 * 32];
  __shared__ alignas(64) _Float16 kbuf[2][32 * 32];   // [key-in-chunk][dh]
  __shared__ alignas(64) _Float16 vbuf[2][32 * 32];   // [dh][t-in-chunk]
  int tid  = threadIdx.x;
  int wid  = tid >> 5;
  int lane = tid & 31;
  int bh   = blockIdx.x >> 4;                 // 16 blocks per (b,h)
  int qt   = (blockIdx.x & 15) * 4 + wid;     // query tile for this wave
  int grp  = lane >> 4, idx = lane & 15;
  const float scale = 0.17677669529663687f;   // 1/sqrt(32)

  const _Float16* Kbase = Kh + (size_t)bh * T_ * DH_;
  const _Float16* Vbase = VhT + (size_t)bh * DH_ * T_;

  // All 128 threads cooperatively stage one 32-key chunk of K and V.
  auto stage = [&](int kc, int bufi) {
    // K chunk: 32 rows x 32 dh = 2KB, contiguous; 16B per thread.
    async_b128_to_lds(Kbase + (size_t)kc * DH_ + tid * 8, &kbuf[bufi][tid * 8]);
    // V chunk: 32 dh-rows x 32 t, each row contiguous (64B = 4 threads).
    int dh = tid >> 2, part = tid & 3;
    async_b128_to_lds(Vbase + (size_t)dh * T_ + kc + part * 8,
                      &vbuf[bufi][dh * 32 + part * 8]);
  };

  const _Float16* qrow = Qh + ((size_t)bh * T_ + qt * 16 + idx) * DH_;
  v16h qa = load_afrag(qrow, 0, grp);

  v8f olo = {}, ohi = {};
  float mrow[8], lrow[8];
#pragma unroll
  for (int r = 0; r < 8; ++r) { mrow[r] = -1e30f; lrow[r] = 0.f; }
  _Float16* pl = pbuf[wid];

  stage(0, 0);
  wait_async0();
  __syncthreads();

  for (int kcix = 0; kcix < T_ / 32; ++kcix) {
    int cur = kcix & 1;
    if (kcix + 1 < T_ / 32) stage((kcix + 1) * 32, 1 - cur);   // overlap DMA

    const _Float16* k1row = &kbuf[cur][idx * 32];
    const _Float16* k2row = &kbuf[cur][(16 + idx) * 32];
    v16h kb1 = load_bfrag(k1row, 0, grp);
    v16h kb2 = load_bfrag(k2row, 0, grp);
    v8f zero = {};
    v8f s1 = __builtin_amdgcn_wmma_f32_16x16x32_f16(false, qa, false, kb1,
                                                    (short)0, zero, false, false);
    v8f s2 = __builtin_amdgcn_wmma_f32_16x16x32_f16(false, qa, false, kb2,
                                                    (short)0, zero, false, false);
    // Online softmax update per row; stage P into LDS (C-layout -> A-layout).
#pragma unroll
    for (int r = 0; r < 8; ++r) {
      float a1 = s1[r] * scale;
      float a2 = s2[r] * scale;
      float mv = fmaxf(a1, a2);
#pragma unroll
      for (int off = 8; off; off >>= 1) mv = fmaxf(mv, __shfl_xor(mv, off, 16));
      float mnew  = fmaxf(mrow[r], mv);
      float alpha = __expf(mrow[r] - mnew);
      float p1 = __expf(a1 - mnew);
      float p2 = __expf(a2 - mnew);
      float sm = p1 + p2;
#pragma unroll
      for (int off = 8; off; off >>= 1) sm += __shfl_xor(sm, off, 16);
      lrow[r] = lrow[r] * alpha + sm;
      mrow[r] = mnew;
      olo[r] *= alpha;
      ohi[r] *= alpha;
      int row = r + grp * 8;
      pl[row * 32 + idx]      = (_Float16)p1;
      pl[row * 32 + 16 + idx] = (_Float16)p2;
    }
    // P as A fragment: two contiguous 8-half LDS loads per lane.
    v16h pa = load_afrag(pl + idx * 32, 0, grp);
    // V fragments from LDS (rows contiguous in t).
    v16h vlo = load_bfrag(&vbuf[cur][idx * 32], 0, grp);
    v16h vhi = load_bfrag(&vbuf[cur][(16 + idx) * 32], 0, grp);
    olo = __builtin_amdgcn_wmma_f32_16x16x32_f16(false, pa, false, vlo,
                                                 (short)0, olo, false, false);
    ohi = __builtin_amdgcn_wmma_f32_16x16x32_f16(false, pa, false, vhi,
                                                 (short)0, ohi, false, false);

    wait_async0();        // next chunk's async copies have landed
    __syncthreads();      // all waves done with 'cur' buffer + see new data
  }

  int b = bh >> 2, h = bh & 3;
#pragma unroll
  for (int r = 0; r < 8; ++r) {
    int row = r + grp * 8;
    int t   = qt * 16 + row;
    float inv = 1.f / lrow[r];
    _Float16* dst = attn16 + ((size_t)b * T_ + t) * D_ + h * DH_;
    dst[idx]      = (_Float16)(olo[r] * inv);
    dst[16 + idx] = (_Float16)(ohi[r] * inv);
  }
}

// ---------------------------------------------------------------------------
// LayerNorm over D=128 per row; optional f32 copy and f16 copy outputs.
// ---------------------------------------------------------------------------
__global__ __launch_bounds__(128)
void ln_kernel(const float* __restrict__ X, const float* __restrict__ w,
               const float* __restrict__ bc, float* __restrict__ Y,
               float* __restrict__ Y2, _Float16* __restrict__ Y16) {
  int wid  = threadIdx.x >> 5;
  int lane = threadIdx.x & 31;
  int row  = blockIdx.x * 4 + wid;
  const float* x = X + (size_t)row * D_;
  float v[4], s = 0.f, ss = 0.f;
#pragma unroll
  for (int i = 0; i < 4; ++i) {
    v[i] = x[lane + 32 * i];
    s  += v[i];
    ss += v[i] * v[i];
  }
#pragma unroll
  for (int off = 16; off; off >>= 1) {
    s  += __shfl_xor(s, off, 32);
    ss += __shfl_xor(ss, off, 32);
  }
  float mean = s * (1.f / 128.f);
  float var  = ss * (1.f / 128.f) - mean * mean;
  float rs   = rsqrtf(var + 1e-5f);
#pragma unroll
  for (int i = 0; i < 4; ++i) {
    int c = lane + 32 * i;
    float val = (v[i] - mean) * rs * w[c] + bc[c];
    if (Y)   Y[(size_t)row * D_ + c]   = val;
    if (Y2)  Y2[(size_t)row * D_ + c]  = val;
    if (Y16) Y16[(size_t)row * D_ + c] = (_Float16)val;
  }
}

// ---------------------------------------------------------------------------
// Cosine-similarity clustering head + softmax over R=4 regimes.
// ---------------------------------------------------------------------------
__global__ __launch_bounds__(128)
void cluster_kernel(const float* __restrict__ hbuf, const float* __restrict__ cent,
                    const float* __restrict__ temp, float* __restrict__ probs) {
  int wid  = threadIdx.x >> 5;
  int lane = threadIdx.x & 31;
  int row  = blockIdx.x * 4 + wid;
  const float* x = hbuf + (size_t)row * D_;
  float hh = 0.f, hc[4] = {0.f, 0.f, 0.f, 0.f}, cc[4] = {0.f, 0.f, 0.f, 0.f};
#pragma unroll
  for (int i = 0; i < 4; ++i) {
    int c = lane + 32 * i;
    float hv = x[c];
    hh += hv * hv;
#pragma unroll
    for (int j = 0; j < 4; ++j) {
      float cv = cent[j * D_ + c];
      hc[j] += hv * cv;
      cc[j] += cv * cv;
    }
  }
#pragma unroll
  for (int off = 16; off; off >>= 1) {
    hh += __shfl_xor(hh, off, 32);
#pragma unroll
    for (int j = 0; j < 4; ++j) {
      hc[j] += __shfl_xor(hc[j], off, 32);
      cc[j] += __shfl_xor(cc[j], off, 32);
    }
  }
  float tinv = 1.f / fmaxf(temp[0], 1e-4f);
  float hn   = fmaxf(sqrtf(hh), 1e-12f);
  float lg[4], mx = -1e30f;
#pragma unroll
  for (int j = 0; j < 4; ++j) {
    lg[j] = hc[j] / (hn * fmaxf(sqrtf(cc[j]), 1e-12f)) * tinv;
    mx = fmaxf(mx, lg[j]);
  }
  float e[4], sum = 0.f;
#pragma unroll
  for (int j = 0; j < 4; ++j) { e[j] = __expf(lg[j] - mx); sum += e[j]; }
  if (lane == 0) {
    float inv = 1.f / sum;
#pragma unroll
    for (int j = 0; j < 4; ++j) probs[(size_t)row * R_ + j] = e[j] * inv;
  }
}

// ---------------------------------------------------------------------------
// Sequential GRU scan + temporal smoothing + argmax. One thread per batch.
// ---------------------------------------------------------------------------
__global__ void gru_kernel(const float* __restrict__ probs, const float* __restrict__ init,
                           const float* __restrict__ wih, const float* __restrict__ whh,
                           const float* __restrict__ bih, const float* __restrict__ bhh,
                           float* __restrict__ smoothed, float* __restrict__ assign,
                           float* __restrict__ finalst) {
  int b = threadIdx.x;
  if (b >= B_) return;
  float Wih[48], Whh[48], Bih[12], Bhh[12];
  for (int i = 0; i < 48; ++i) { Wih[i] = wih[i]; Whh[i] = whh[i]; }
  for (int i = 0; i < 12; ++i) { Bih[i] = bih[i]; Bhh[i] = bhh[i]; }
  float st[4];
  for (int j = 0; j < 4; ++j) st[j] = init[b * R_ + j];

  for (int t = 0; t < T_; ++t) {
    const float* p = probs + ((size_t)b * T_ + t) * R_;
    float pv[4] = {p[0], p[1], p[2], p[3]};
    float gi[12], gh[12];
#pragma unroll
    for (int i = 0; i < 12; ++i) {
      float a = Bih[i], c = Bhh[i];
#pragma unroll
      for (int j = 0; j < 4; ++j) {
        a += Wih[i * 4 + j] * pv[j];
        c += Whh[i * 4 + j] * st[j];
      }
      gi[i] = a; gh[i] = c;
    }
    float ns[4];
#pragma unroll
    for (int j = 0; j < 4; ++j) {
      float r = 1.f / (1.f + __expf(-(gi[j] + gh[j])));
      float z = 1.f / (1.f + __expf(-(gi[4 + j] + gh[4 + j])));
      float n = tanhf(gi[8 + j] + r * gh[8 + j]);
      float s = (1.f - z) * n + z * st[j];
      ns[j] = s * 0.9f + pv[j] * 0.1f;    // temporal smoothing TS=0.9
    }
    int best = 0; float bv = ns[0];
#pragma unroll
    for (int j = 1; j < 4; ++j) if (ns[j] > bv) { bv = ns[j]; best = j; }
    float* so = smoothed + ((size_t)b * T_ + t) * R_;
#pragma unroll
    for (int j = 0; j < 4; ++j) { so[j] = ns[j]; st[j] = ns[j]; }
    assign[(size_t)b * T_ + t] = (float)best;
  }
  for (int j = 0; j < 4; ++j) finalst[b * R_ + j] = st[j];
}

// ---------------------------------------------------------------------------
extern "C" void kernel_launch(void* const* d_in, const int* in_sizes, int n_in,
                              void* d_out, int out_size, void* d_ws, size_t ws_size,
                              hipStream_t stream) {
  const float* x     = (const float*)d_in[0];
  const float* init  = (const float*)d_in[1];
  const float* in_w  = (const float*)d_in[2];
  const float* in_b  = (const float*)d_in[3];
  const float* qkv_w = (const float*)d_in[4];
  const float* qkv_b = (const float*)d_in[5];
  const float* out_w = (const float*)d_in[6];
  const float* out_b = (const float*)d_in[7];
  const float* ln1w  = (const float*)d_in[8];
  const float* ln1b  = (const float*)d_in[9];
  const float* ff1w  = (const float*)d_in[10];
  const float* ff1b  = (const float*)d_in[11];
  const float* ff2w  = (const float*)d_in[12];
  const float* ff2b  = (const float*)d_in[13];
  const float* ln2w  = (const float*)d_in[14];
  const float* ln2b  = (const float*)d_in[15];
  const float* nw    = (const float*)d_in[16];
  const float* nb    = (const float*)d_in[17];
  const float* cent  = (const float*)d_in[18];
  const float* temp  = (const float*)d_in[19];
  const float* gwih  = (const float*)d_in[20];
  const float* gwhh  = (const float*)d_in[21];
  const float* gbih  = (const float*)d_in[22];
  const float* gbhh  = (const float*)d_in[23];

  float* out_sm  = (float*)d_out;                     // [B,T,R]
  float* out_asn = out_sm + (size_t)BT_ * R_;         // [B,T]
  float* out_pr  = out_asn + (size_t)BT_;             // [B,T,R]
  float* out_h   = out_pr + (size_t)BT_ * R_;         // [B,T,D]
  float* out_fs  = out_h + (size_t)BT_ * D_;          // [B,R]

  // fp32 workspace
  float* ws_h   = (float*)d_ws;                       // [BT,D] residual master
  float* ws_tmp = ws_h + (size_t)BT_ * D_;            // [BT,D] pre-LN sums
  // f16 workspace
  _Float16* p16   = (_Float16*)(ws_tmp + (size_t)BT_ * D_);
  _Float16* h16   = p16;  p16 += (size_t)BT_ * D_;    // activations f16
  _Float16* at16  = p16;  p16 += (size_t)BT_ * D_;    // attention out f16
  _Float16* ff16  = p16;  p16 += (size_t)BT_ * DFF_;  // relu(ff1) f16
  _Float16* x16   = p16;  p16 += (size_t)BT_ * NF_;   // input f16
  _Float16* q16   = p16;  p16 += (size_t)BT_ * D_;
  _Float16* k16   = p16;  p16 += (size_t)BT_ * D_;
  _Float16* v16   = p16;  p16 += (size_t)BT_ * D_;
  _Float16* w_in  = p16;  p16 += (size_t)D_ * NF_;
  _Float16* w_qkv = p16;  p16 += (size_t)L_ * 3 * D_ * D_;
  _Float16* w_out = p16;  p16 += (size_t)L_ * D_ * D_;
  _Float16* w_f1  = p16;  p16 += (size_t)L_ * DFF_ * D_;
  _Float16* w_f2  = p16;  p16 += (size_t)L_ * D_ * DFF_;

  auto pack = [&](const float* s, _Float16* d, int n) {
    pack_f16<<<(n + 255) / 256, 256, 0, stream>>>(s, d, n);
  };
  pack(x,     x16,   BT_ * NF_);
  pack(in_w,  w_in,  D_ * NF_);
  pack(qkv_w, w_qkv, L_ * 3 * D_ * D_);
  pack(out_w, w_out, L_ * D_ * D_);
  pack(ff1w,  w_f1,  L_ * DFF_ * D_);
  pack(ff2w,  w_f2,  L_ * D_ * DFF_);

  // Input projection: h = x @ in_w^T + in_b  (f32 + f16 outputs)
  {
    int tiles = (BT_ / 16) * (D_ / 16);
    gemm16<<<(tiles + 3) / 4, 128, 0, stream>>>(x16, w_in, in_b, nullptr,
                                                ws_h, h16, BT_, D_, NF_, 0);
  }

  for (int l = 0; l < L_; ++l) {
    int tq = (BT_ / 16) * ((3 * D_) / 16);
    gemm_qkv16<<<(tq + 3) / 4, 128, 0, stream>>>(
        h16, w_qkv + (size_t)l * 3 * D_ * D_, qkv_b + (size_t)l * 3 * D_,
        q16, k16, v16);

    attn_kernel<<<B_ * H_ * (T_ / 16) / 4, 128, 0, stream>>>(q16, k16, v16, at16);

    int to = (BT_ / 16) * (D_ / 16);
    gemm16<<<(to + 3) / 4, 128, 0, stream>>>(
        at16, w_out + (size_t)l * D_ * D_, out_b + (size_t)l * D_, ws_h,
        ws_tmp, nullptr, BT_, D_, D_, 2);
    ln_kernel<<<BT_ / 4, 128, 0, stream>>>(ws_tmp, ln1w + l * D_, ln1b + l * D_,
                                           ws_h, nullptr, h16);

    int t1 = (BT_ / 16) * (DFF_ / 16);
    gemm16<<<(t1 + 3) / 4, 128, 0, stream>>>(
        h16, w_f1 + (size_t)l * DFF_ * D_, ff1b + (size_t)l * DFF_, nullptr,
        nullptr, ff16, BT_, DFF_, D_, 1);

    int t2 = (BT_ / 16) * (D_ / 16);
    gemm16<<<(t2 + 3) / 4, 128, 0, stream>>>(
        ff16, w_f2 + (size_t)l * D_ * DFF_, ff2b + (size_t)l * D_, ws_h,
        ws_tmp, nullptr, BT_, D_, DFF_, 2);
    ln_kernel<<<BT_ / 4, 128, 0, stream>>>(ws_tmp, ln2w + l * D_, ln2b + l * D_,
                                           ws_h, nullptr, h16);
  }

  // Final norm (writes h output too), clustering, GRU scan.
  ln_kernel<<<BT_ / 4, 128, 0, stream>>>(ws_h, nw, nb, ws_tmp, out_h, nullptr);
  cluster_kernel<<<BT_ / 4, 128, 0, stream>>>(ws_tmp, cent, temp, out_pr);
  gru_kernel<<<1, 32, 0, stream>>>(out_pr, init, gwih, gwhh, gbih, gbhh,
                                   out_sm, out_asn, out_fs);
}
